// gat_net_35261681500391
// MI455X (gfx1250) — matrix operand order repaired and checked by reading, not compiled, not run on hardware
//
#include <hip/hip_runtime.h>
#include <cstdint>
#include <cstddef>

typedef __attribute__((ext_vector_type(16))) _Float16 v16h;
typedef __attribute__((ext_vector_type(8)))  _Float16 v8h;
typedef __attribute__((ext_vector_type(8)))  float    v8f;

#define AST 136   // f16 row stride for 128-wide activation tiles (272B rows, 16B aligned, bank-skewed)
#define HST 264   // f16 row stride for 256-wide ffn-hidden tiles

// ---------------------------------------------------------------------------
// Weight repack: f32 W[K][Nout] row-major -> WMMA B fragments
// layout: [kt][nt][lane][kk] ; lane&15 = N column, lane>>4 selects K half,
// per-lane K values: kk<8 -> K half*8+kk, kk>=8 -> K 16+half*8+(kk-8) (matches A fragment order).
// ---------------------------------------------------------------------------
__global__ __launch_bounds__(256) void pack_b_frags(const float* __restrict__ W,
                                                    _Float16* __restrict__ out,
                                                    int K, int Nout) {
  int idx = blockIdx.x * 256 + threadIdx.x;
  int total = K * Nout;
  if (idx >= total) return;
  int NT = Nout >> 4;
  int kk = idx & 15;
  int t = idx >> 4;
  int lane = t & 31; t >>= 5;
  int nt = t % NT;
  int kt = t / NT;
  int krow = kt * 32 + (lane >> 4) * 8 + (kk & 7) + (kk >> 3) * 16;
  int ncol = nt * 16 + (lane & 15);
  out[idx] = (_Float16)W[(size_t)krow * Nout + ncol];
}

// ---------------------------------------------------------------------------
// Fragment helpers
// ---------------------------------------------------------------------------
__device__ __forceinline__ v16h load_a_frag(const _Float16* buf, int stride, int m0, int k0) {
  int lane = threadIdx.x & 31;
  int m = m0 + (lane & 15);
  int hi = lane >> 4;
  const _Float16* p = buf + m * stride + k0 + hi * 8;   // K runs [k0+hi*8,+8) and [k0+16+hi*8,+8)
  v8h lo = *(const v8h*)p;
  v8h hh = *(const v8h*)(p + 16);
  v16h r;
#pragma unroll
  for (int i = 0; i < 8; ++i) { r[i] = lo[i]; r[i + 8] = hh[i]; }
  return r;
}

__device__ __forceinline__ v16h load_b_frag(const _Float16* frags, int ntiles, int kt, int nt) {
  int lane = threadIdx.x & 31;
  return *(const v16h*)(frags + (size_t)(((kt * ntiles + nt) * 32) + lane) * 16);
}

__device__ __forceinline__ v8f wmma16(v16h a, v16h b, v8f c) {
  return __builtin_amdgcn_wmma_f32_16x16x32_f16(false, a, false, b, (short)0, c, false, false);
}

__device__ __forceinline__ void store_c_f16(_Float16* buf, int stride, int m0, int n0, v8f c) {
  int lane = threadIdx.x & 31;
  int n = n0 + (lane & 15);
  int hi = lane >> 4;
#pragma unroll
  for (int r = 0; r < 8; ++r)
    buf[(m0 + r + 8 * hi) * stride + n] = (_Float16)c[r];  // C layout: vgpr r -> row r+8*hi
}

// wave w handles rows 2w (lanes 0-15) and 2w+1 (lanes 16-31); 8 cols/lane; shuffle reduce.
__device__ __forceinline__ void layer_norm_16x128(const _Float16* src, _Float16* dst,
                                                  const float* __restrict__ gamma,
                                                  const float* __restrict__ beta) {
  int lane = threadIdx.x & 31, wave = threadIdx.x >> 5;
  int row = wave * 2 + (lane >> 4);
  int c0 = (lane & 15) * 8;
  const _Float16* rp = src + row * AST;
  float v[8], s = 0.f, q = 0.f;
#pragma unroll
  for (int e = 0; e < 8; ++e) { float x = (float)rp[c0 + e]; v[e] = x; s += x; q += x * x; }
#pragma unroll
  for (int o = 1; o < 16; o <<= 1) { s += __shfl_xor(s, o, 32); q += __shfl_xor(q, o, 32); }
  float mean = s * (1.f / 128.f);
  float var = q * (1.f / 128.f) - mean * mean;
  float rinv = rsqrtf(var + 1e-5f);
  _Float16* wp = dst + row * AST;
#pragma unroll
  for (int e = 0; e < 8; ++e) {
    int cc = c0 + e;
    wp[cc] = (_Float16)((v[e] - mean) * rinv * gamma[cc] + beta[cc]);
  }
}

// ---------------------------------------------------------------------------
struct GatParams {
  const float* feat;
  const int* nidx;
  const unsigned char* nvalid;   // jnp bool -> 1 byte
  const _Float16* wq[2]; const _Float16* wk[2]; const _Float16* wv[2]; const _Float16* wo[2];
  const _Float16* w1[2]; const _Float16* w2[2]; const _Float16* w4;
  const float* bo[2]; const float* b1[2]; const float* b2[2];
  const float* ga[2]; const float* ba[2]; const float* gb[2]; const float* bb[2];
  const float* b4;
  float* out;
  int n;
};

__global__ __launch_bounds__(256, 1) void gat_fused(GatParams p) {
  const int tid  = threadIdx.x;
  const int lane = tid & 31;
  const int wave = tid >> 5;
  const int hi   = lane >> 4;
  const int ln   = lane & 15;
  const int node0 = blockIdx.x * 16;

  __shared__ __align__(16) _Float16 sK[80 * AST];     // raw masked neighbor rows (both layers)
  __shared__ __align__(16) _Float16 sIn[16 * AST];    // layer input (q rows)
  __shared__ __align__(16) _Float16 sQp[16 * AST];    // projected Q
  __shared__ __align__(16) _Float16 sReg[80 * AST];   // aliased: K' | kb+attnOut | x+ffnHidden
  __shared__ float sAw[16 * 8 * 5];                   // attention weights
  __shared__ unsigned sMaskSh[16];

  _Float16* sKp = sReg;                 // 80 x AST
  _Float16* sKB = sReg;                 // 64 x AST (weighted neighbor rows, per head group)
  _Float16* sAt = sReg + 64 * AST;      // 16 x AST (attention output)
  _Float16* sXb = sReg;                 // 16 x AST (pre/post-LN x)
  _Float16* sHb = sReg + 16 * AST;      // 16 x HST (ffn hidden)

  // next-block prefetch (emits global_prefetch_b8)
  if (tid < 16) {
    int nb = node0 + 16 + tid;
    if (nb < p.n) __builtin_prefetch(&p.feat[(size_t)nb * 128], 0, 0);
  }

  // ---- gather: q rows + masked neighbor rows -> LDS f16 ----
  for (int e = tid; e < 16 * 128; e += 256) {
    int r = e >> 7, c = e & 127;
    int rg = node0 + r; if (rg >= p.n) rg = p.n - 1;
    sIn[r * AST + c] = (_Float16)p.feat[(size_t)rg * 128 + c];
  }
  if (tid < 16) {
    int rg = node0 + tid; if (rg >= p.n) rg = p.n - 1;
    unsigned mb = 0;
    for (int j = 0; j < 5; ++j)
      if (!p.nvalid[rg * 5 + j]) mb |= 1u << j;
    sMaskSh[tid] = mb;
  }
  for (int e = tid; e < 80 * 128; e += 256) {
    int r = e >> 7, c = e & 127;
    int i = r / 5, j = r - i * 5;
    int rg = node0 + i; if (rg >= p.n) rg = p.n - 1;
    int base = rg * 5 + j;
    float v = 0.f;
    if (p.nvalid[base]) v = p.feat[(size_t)p.nidx[base] * 128 + c];
    sK[r * AST + c] = (_Float16)v;
  }
  __syncthreads();

  for (int layer = 0; layer < 2; ++layer) {
    // ---- Q' = in @ Wq (16x128 @ 128x128); wave = n-tile ----
    {
      v8f c = {};
#pragma unroll
      for (int kt = 0; kt < 4; ++kt)
        c = wmma16(load_a_frag(sIn, AST, 0, kt * 32), load_b_frag(p.wq[layer], 8, kt, wave), c);
      store_c_f16(sQp, AST, 0, wave * 16, c);
    }
    // ---- K' = k @ Wk (80x128 @ 128x128); 5 m-tiles per wave ----
    for (int mt = 0; mt < 5; ++mt) {
      v8f c = {};
#pragma unroll
      for (int kt = 0; kt < 4; ++kt)
        c = wmma16(load_a_frag(sK, AST, mt * 16, kt * 32), load_b_frag(p.wk[layer], 8, kt, wave), c);
      store_c_f16(sKp, AST, mt * 16, wave * 16, c);
    }
    __syncthreads();

    // ---- attention weights: thread (i,h), 5 dots of length 16, masked softmax ----
    if (tid < 128) {
      int i = tid >> 3, h = tid & 7;
      const _Float16* qr = sQp + i * AST + h * 16;
      unsigned mb = sMaskSh[i];
      float s[5], mx = -3.0e38f;
      for (int j = 0; j < 5; ++j) {
        const _Float16* kr = sKp + (i * 5 + j) * AST + h * 16;
        float d = 0.f;
#pragma unroll
        for (int t = 0; t < 16; ++t) d += (float)qr[t] * (float)kr[t];
        d *= 0.25f;                      // dh^-0.5, dh=16
        if (mb & (1u << j)) d = -3.0e38f;
        s[j] = d; mx = fmaxf(mx, d);
      }
      float se = 0.f;
      for (int j = 0; j < 5; ++j) { s[j] = __expf(s[j] - mx); se += s[j]; }
      float inv = 1.f / se;
      for (int j = 0; j < 5; ++j) sAw[(i * 8 + h) * 5 + j] = s[j] * inv;
    }
    __syncthreads();   // sQp/sKp dead -> region reusable

    // ---- attn@V via linearity: kb = attn-weighted raw rows, then kb @ Wv ----
    for (int g = 0; g < 2; ++g) {
      {  // build kb for heads g*4..g*4+3 : 64 rows x 128; 4 threads/row, 32 cols each
        int row = tid >> 2;                 // 0..63
        int hh = row >> 4, i = row & 15;
        int h = g * 4 + hh;
        int c0 = (tid & 3) * 32;
        const float* aw = &sAw[(i * 8 + h) * 5];
        float a0 = aw[0], a1 = aw[1], a2 = aw[2], a3 = aw[3], a4 = aw[4];
        const _Float16* k0 = sK + (i * 5 + 0) * AST;
        const _Float16* k1 = sK + (i * 5 + 1) * AST;
        const _Float16* k2 = sK + (i * 5 + 2) * AST;
        const _Float16* k3 = sK + (i * 5 + 3) * AST;
        const _Float16* k4 = sK + (i * 5 + 4) * AST;
        _Float16* kb = sKB + (hh * 16 + i) * AST;
#pragma unroll 8
        for (int c = c0; c < c0 + 32; ++c) {
          float v = a0 * (float)k0[c] + a1 * (float)k1[c] + a2 * (float)k2[c] +
                    a3 * (float)k3[c] + a4 * (float)k4[c];
          kb[c] = (_Float16)v;
        }
      }
      __syncthreads();
      if (wave < 4) {       // wave = head-in-group; output columns = head slice of Wv
        int h = g * 4 + wave;
        v8f c = {};
#pragma unroll
        for (int kt = 0; kt < 4; ++kt)
          c = wmma16(load_a_frag(sKB, AST, wave * 16, kt * 32), load_b_frag(p.wv[layer], 8, kt, h), c);
        store_c_f16(sAt, AST, 0, h * 16, c);
      }
      __syncthreads();
    }

    // ---- O = attnOut @ Wo + bo, + residual(in) -> pre-LN x ----
    {
      int n = wave * 16 + ln;
      float bias = p.bo[layer][n];
      v8f c;
#pragma unroll
      for (int r = 0; r < 8; ++r) c[r] = bias;
#pragma unroll
      for (int kt = 0; kt < 4; ++kt)
        c = wmma16(load_a_frag(sAt, AST, 0, kt * 32), load_b_frag(p.wo[layer], 8, kt, wave), c);
#pragma unroll
      for (int r = 0; r < 8; ++r) {
        int m = r + 8 * hi;
        sXb[m * AST + n] = (_Float16)(c[r] + (float)sIn[m * AST + n]);
      }
    }
    __syncthreads();
    layer_norm_16x128(sXb, sXb, p.ga[layer], p.ba[layer]);   // x = LN1(q+out), in place
    __syncthreads();

    // ---- FFN1: H = relu(x @ W1 + b1), 16 n-tiles (2 per wave) ----
#pragma unroll
    for (int half = 0; half < 2; ++half) {
      int nt = wave + half * 8;
      int n = nt * 16 + ln;
      float bias = p.b1[layer][n];
      v8f c;
#pragma unroll
      for (int r = 0; r < 8; ++r) c[r] = bias;
#pragma unroll
      for (int kt = 0; kt < 4; ++kt)
        c = wmma16(load_a_frag(sXb, AST, 0, kt * 32), load_b_frag(p.w1[layer], 16, kt, nt), c);
#pragma unroll
      for (int r = 0; r < 8; ++r) c[r] = fmaxf(c[r], 0.f);
      store_c_f16(sHb, HST, 0, nt * 16, c);
    }
    __syncthreads();

    // ---- FFN2: F = H @ W2 + b2, + residual(x) -> pre-LN2, in place ----
    {
      int n = wave * 16 + ln;
      float bias = p.b2[layer][n];
      v8f c;
#pragma unroll
      for (int r = 0; r < 8; ++r) c[r] = bias;
#pragma unroll
      for (int kt = 0; kt < 8; ++kt)
        c = wmma16(load_a_frag(sHb, HST, 0, kt * 32), load_b_frag(p.w2[layer], 8, kt, wave), c);
#pragma unroll
      for (int r = 0; r < 8; ++r) {
        int m = r + 8 * hi;
        sXb[m * AST + n] = (_Float16)(c[r] + (float)sXb[m * AST + n]);
      }
    }
    __syncthreads();
    layer_norm_16x128(sXb, sIn, p.gb[layer], p.bb[layer]);   // next-layer input
    __syncthreads();
  }

  // ---- head: out = tanh(in @ W4 + b4), 4 n-tiles on waves 0-3 ----
  if (wave < 4) {
    int n = wave * 16 + ln;
    float bias = p.b4[n];
    v8f c;
#pragma unroll
    for (int r = 0; r < 8; ++r) c[r] = bias;
#pragma unroll
    for (int kt = 0; kt < 4; ++kt)
      c = wmma16(load_a_frag(sIn, AST, 0, kt * 32), load_b_frag(p.w4, 4, kt, wave), c);
#pragma unroll
    for (int r = 0; r < 8; ++r) {
      int m = node0 + r + 8 * hi;
      if (m < p.n) p.out[(size_t)m * 64 + n] = tanhf(c[r]);
    }
  }
}

// ---------------------------------------------------------------------------
extern "C" void kernel_launch(void* const* d_in, const int* in_sizes, int n_in,
                              void* d_out, int out_size, void* d_ws, size_t ws_size,
                              hipStream_t stream) {
  const int N = in_sizes[0] / 128;
  _Float16* wsf = (_Float16*)d_ws;
  if (ws_size < (size_t)270336 * sizeof(_Float16)) return;   // packed-weight scratch

  // packed fragment offsets (f16 elements)
  const size_t OFF[13] = {0, 16384, 32768, 49152, 65536, 98304,
                          131072, 147456, 163840, 180224, 196608, 229376, 262144};
  // source input index, K, Nout for each matrix
  const int SRC[13] = {3, 4, 5, 6, 8, 10, 16, 17, 18, 19, 21, 23, 29};
  const int KD [13] = {128,128,128,128,128,256, 128,128,128,128,128,256, 128};
  const int ND [13] = {128,128,128,128,256,128, 128,128,128,128,256,128, 64};
  for (int m = 0; m < 13; ++m) {
    int total = KD[m] * ND[m];
    pack_b_frags<<<(total + 255) / 256, 256, 0, stream>>>(
        (const float*)d_in[SRC[m]], wsf + OFF[m], KD[m], ND[m]);
  }

  GatParams p;
  p.feat   = (const float*)d_in[0];
  p.nidx   = (const int*)d_in[1];
  p.nvalid = (const unsigned char*)d_in[2];
  for (int l = 0; l < 2; ++l) {
    int b = 3 + l * 13, f = l * 6;
    p.wq[l] = wsf + OFF[f + 0];
    p.wk[l] = wsf + OFF[f + 1];
    p.wv[l] = wsf + OFF[f + 2];
    p.wo[l] = wsf + OFF[f + 3];
    p.w1[l] = wsf + OFF[f + 4];
    p.w2[l] = wsf + OFF[f + 5];
    p.bo[l] = (const float*)d_in[b + 4];
    p.b1[l] = (const float*)d_in[b + 6];
    p.b2[l] = (const float*)d_in[b + 8];
    p.ga[l] = (const float*)d_in[b + 9];
    p.ba[l] = (const float*)d_in[b + 10];
    p.gb[l] = (const float*)d_in[b + 11];
    p.bb[l] = (const float*)d_in[b + 12];
  }
  p.w4  = wsf + OFF[12];
  p.b4  = (const float*)d_in[30];
  p.out = (float*)d_out;
  p.n   = N;

  int blocks = (N + 15) / 16;
  gat_fused<<<blocks, 256, 0, stream>>>(p);
}
